// Encoder_50723563766556
// MI455X (gfx1250) — matrix operand order repaired and checked by reading, not compiled, not run on hardware
//
#include <hip/hip_runtime.h>
#include <hip/hip_bf16.h>

typedef _Float16 half8 __attribute__((ext_vector_type(8)));
typedef _Float16 v16h  __attribute__((ext_vector_type(16)));
typedef float    v8f   __attribute__((ext_vector_type(8)));

#define GAT_N 4096
#define GAT_F 512
#define GAT_H 256
#define NEGV  (-9.0e15f)
#define BNEPS 1e-5f

// ---------------- f32 -> f16 convert (grid-stride) ----------------
__global__ __launch_bounds__(256) void k_cvt_f16(
    _Float16* __restrict__ dst, const float* __restrict__ src, int n) {
  for (int i = blockIdx.x * 256 + threadIdx.x; i < n; i += gridDim.x * 256)
    dst[i] = (_Float16)src[i];
}

// ------------- f32 [rows][cols] -> f16 [cols][rows] transpose -------------
__global__ __launch_bounds__(256) void k_transpose_f16(
    _Float16* __restrict__ dst, const float* __restrict__ src,
    int rows, int cols) {
  int n = rows * cols;
  for (int i = blockIdx.x * 256 + threadIdx.x; i < n; i += gridDim.x * 256) {
    int r = i / cols, c = i - r * cols;
    dst[(size_t)c * rows + r] = (_Float16)src[i];
  }
}

// A-operand fragment: lane holds row M=lane%16; halves K=kbA..kbA+7 and
// kbA+16..kbA+23 (kbA = 0 / 8 per lane half) -> byte offsets 0 and +32.
__device__ __forceinline__ v16h load_afrag(const _Float16* p) {
  const half8* q = (const half8*)p;
  half8 lo = q[0], hi = q[2];
  v16h v;
#pragma unroll
  for (int i = 0; i < 8; ++i) { v[i] = lo[i]; v[8 + i] = hi[i]; }
  return v;
}

// B-operand fragment: lane holds col N=lane%16; halves K=kbB..kbB+7 and
// kbB+16..kbB+23 (kbB = 0 / 8 per lane half) -> byte offsets 0 and +32.
__device__ __forceinline__ v16h load_bfrag(const _Float16* p) {
  const half8* q = (const half8*)p;
  half8 lo = q[0], hi = q[2];
  v16h v;
#pragma unroll
  for (int i = 0; i < 8; ++i) { v[i] = lo[i]; v[8 + i] = hi[i]; }
  return v;
}

// ---------------- WMMA GEMM: C[M,N] = A[M,K] * Bt[N,K]^T ----------------
// A row-major f16, Bt row-major f16 (B transposed), C row-major f32.
// One wave computes a 32x32 C tile as 2x2 WMMA tiles (A/B fragments reused
// across two wmmas each): 8 b128 loads feed 4 v_wmma per K-step of 32.
__global__ __launch_bounds__(256) void k_gemm32(
    const _Float16* __restrict__ A, const _Float16* __restrict__ Bt,
    float* __restrict__ C, int M, int Nn, int K, int relu) {
  int wave = (blockIdx.x * 256 + threadIdx.x) >> 5;
  int lane = threadIdx.x & 31;
  int nwav = (gridDim.x * 256) >> 5;
  int mt = M >> 5, nt = Nn >> 5;
  int ntiles = mt * nt;
  int r   = lane & 15;
  int kbA = (lane < 16) ? 0 : 8;
  int kbB = (lane < 16) ? 0 : 8;
  int rowoff = (lane < 16) ? 0 : 8;

  for (int t = wave; t < ntiles; t += nwav) {
    int tm = (t / nt) << 5;
    int tn = (t - (t / nt) * nt) << 5;
    const _Float16* arow0 = A + (size_t)(tm + r) * K + kbA;
    const _Float16* arow1 = arow0 + (size_t)16 * K;
    const _Float16* brow0 = Bt + (size_t)(tn + r) * K + kbB;
    const _Float16* brow1 = brow0 + (size_t)16 * K;
    v8f c00 = {0.f, 0.f, 0.f, 0.f, 0.f, 0.f, 0.f, 0.f};
    v8f c01 = c00, c10 = c00, c11 = c00;
    for (int k = 0; k < K; k += 32) {
      v16h a0 = load_afrag(arow0 + k);
      v16h a1 = load_afrag(arow1 + k);
      v16h b0 = load_bfrag(brow0 + k);
      v16h b1 = load_bfrag(brow1 + k);
      c00 = __builtin_amdgcn_wmma_f32_16x16x32_f16(false, a0, false, b0,
                                                   (short)0, c00, false, false);
      c01 = __builtin_amdgcn_wmma_f32_16x16x32_f16(false, a0, false, b1,
                                                   (short)0, c01, false, false);
      c10 = __builtin_amdgcn_wmma_f32_16x16x32_f16(false, a1, false, b0,
                                                   (short)0, c10, false, false);
      c11 = __builtin_amdgcn_wmma_f32_16x16x32_f16(false, a1, false, b1,
                                                   (short)0, c11, false, false);
    }
    // C/D layout: VGPR p -> row m0 + p + (lane<16?0:8), col n0 + (lane&15)
#pragma unroll
    for (int sub = 0; sub < 4; ++sub) {
      v8f cc = (sub == 0) ? c00 : (sub == 1) ? c01 : (sub == 2) ? c10 : c11;
      int m0 = tm + ((sub >> 1) << 4);
      int n0 = tn + ((sub & 1) << 4);
      float* crow = C + (size_t)(m0 + rowoff) * Nn + n0 + r;
#pragma unroll
      for (int p = 0; p < 8; ++p) {
        float v = cc[p];
        if (relu) v = v > 0.f ? v : 0.f;
        crow[(size_t)p * Nn] = v;
      }
    }
  }
}

// ------------- per-row dual dot: s[i]=Wh[i,:]@asrc, d[i]=Wh[i,:]@adst -------------
__global__ __launch_bounds__(256) void k_rowdot2(
    const float* __restrict__ Wh, const float* __restrict__ asrc,
    const float* __restrict__ adst, float* __restrict__ s,
    float* __restrict__ d, int Hh) {
  __shared__ float r1[256], r2[256];
  int row = blockIdx.x, t = threadIdx.x;
  float x = (t < Hh) ? Wh[(size_t)row * Hh + t] : 0.f;
  r1[t] = (t < Hh) ? x * asrc[t] : 0.f;
  r2[t] = (t < Hh) ? x * adst[t] : 0.f;
  __syncthreads();
  for (int off = 128; off; off >>= 1) {
    if (t < off) { r1[t] += r1[t + off]; r2[t] += r2[t + off]; }
    __syncthreads();
  }
  if (t == 0) { s[row] = r1[0]; d[row] = r2[0]; }
}

// -------- masked softmax attention row: P[i,:] (f16), one block per row --------
__global__ __launch_bounds__(256) void k_attn_row(
    const float* __restrict__ s, const float* __restrict__ dvec,
    const int* __restrict__ adj, _Float16* __restrict__ P) {
  __shared__ float red[256];
  const int row = blockIdx.x, t = threadIdx.x;
  const float si = s[row];
  const int* arow = adj + (size_t)row * GAT_N;
  float vals[GAT_N / 256];
  float lmax = -3.0e38f;
#pragma unroll
  for (int c = 0; c < GAT_N / 256; ++c) {
    int j = t + c * 256;
    float e = si + dvec[j];
    e = e > 0.f ? e : 0.2f * e;           // leaky_relu(., 0.2)
    float v = (arow[j] != 0) ? e : NEGV;  // where(adj, e, NEG)
    vals[c] = v;
    lmax = fmaxf(lmax, v);
  }
  red[t] = lmax; __syncthreads();
  for (int off = 128; off; off >>= 1) {
    if (t < off) red[t] = fmaxf(red[t], red[t + off]);
    __syncthreads();
  }
  float rmax = red[0]; __syncthreads();
  float lsum = 0.f;
#pragma unroll
  for (int c = 0; c < GAT_N / 256; ++c) {
    vals[c] = __expf(vals[c] - rmax);     // masked -> exp(-huge) = 0
    lsum += vals[c];
  }
  red[t] = lsum; __syncthreads();
  for (int off = 128; off; off >>= 1) {
    if (t < off) red[t] += red[t + off];
    __syncthreads();
  }
  float inv = 1.0f / red[0];
  _Float16* prow = P + (size_t)row * GAT_N;
#pragma unroll
  for (int c = 0; c < GAT_N / 256; ++c) {
    int j = t + c * 256;
    prow[j] = (_Float16)(vals[c] * inv);
  }
}

// --------- per-feature batch stats: mu, rstd over node axis (biased var) ---------
__global__ __launch_bounds__(256) void k_colstats(
    const float* __restrict__ X, float* __restrict__ mu,
    float* __restrict__ rstd, int Nn, int Hh) {
  __shared__ float r1[256], r2[256];
  int h = blockIdx.x, t = threadIdx.x;
  float sum = 0.f, sq = 0.f;
  for (int i = t; i < Nn; i += 256) {
    float v = X[(size_t)i * Hh + h];
    sum += v; sq += v * v;
  }
  r1[t] = sum; r2[t] = sq; __syncthreads();
  for (int off = 128; off; off >>= 1) {
    if (t < off) { r1[t] += r1[t + off]; r2[t] += r2[t + off]; }
    __syncthreads();
  }
  if (t == 0) {
    float m = r1[0] / (float)Nn;
    float var = r2[0] / (float)Nn - m * m;
    mu[h] = m;
    rstd[h] = rsqrtf(var + BNEPS);
  }
}

// ---------------- BN apply (f16 out for next layer / f32 out final) ----------------
__global__ __launch_bounds__(256) void k_bn_f16(
    const float* __restrict__ X, const float* __restrict__ mu,
    const float* __restrict__ rstd, const float* __restrict__ gamma,
    const float* __restrict__ beta, _Float16* __restrict__ out,
    int n, int Hh) {
  for (int i = blockIdx.x * 256 + threadIdx.x; i < n; i += gridDim.x * 256) {
    int h = i & (Hh - 1);
    out[i] = (_Float16)(gamma[h] * (X[i] - mu[h]) * rstd[h] + beta[h]);
  }
}

__global__ __launch_bounds__(256) void k_bn_f32(
    const float* __restrict__ X, const float* __restrict__ mu,
    const float* __restrict__ rstd, const float* __restrict__ gamma,
    const float* __restrict__ beta, float* __restrict__ out,
    int n, int Hh) {
  for (int i = blockIdx.x * 256 + threadIdx.x; i < n; i += gridDim.x * 256) {
    int h = i & (Hh - 1);
    out[i] = gamma[h] * (X[i] - mu[h]) * rstd[h] + beta[h];
  }
}

extern "C" void kernel_launch(void* const* d_in, const int* in_sizes, int n_in,
                              void* d_out, int out_size, void* d_ws, size_t ws_size,
                              hipStream_t stream) {
  (void)in_sizes; (void)n_in; (void)out_size; (void)ws_size;
  const float* x      = (const float*)d_in[0];
  const int*   adj    = (const int*)  d_in[1];
  const float* W1     = (const float*)d_in[2];
  const float* a1s    = (const float*)d_in[3];
  const float* a1d    = (const float*)d_in[4];
  const float* gamma1 = (const float*)d_in[5];
  const float* beta1  = (const float*)d_in[6];
  const float* W2     = (const float*)d_in[7];
  const float* a2s    = (const float*)d_in[8];
  const float* a2d    = (const float*)d_in[9];
  const float* gamma2 = (const float*)d_in[10];
  const float* beta2  = (const float*)d_in[11];
  float* out = (float*)d_out;

  const int N = GAT_N, F = GAT_F, H = GAT_H;

  // ---- workspace carve-up (256B aligned) ----
  char* ws = (char*)d_ws;
  size_t off = 0;
  auto carve = [&](size_t bytes) -> char* {
    char* p = ws + off;
    off = (off + bytes + 255) & ~(size_t)255;
    return p;
  };
  _Float16* XH   = (_Float16*)carve((size_t)N * F * 2);  // x in f16
  _Float16* W1T  = (_Float16*)carve((size_t)H * F * 2);  // W1^T f16  [H][F]
  _Float16* W2T  = (_Float16*)carve((size_t)H * H * 2);  // W2^T f16  [H][H]
  float*    WH   = (float*)   carve((size_t)N * H * 4);  // Wh f32    [N][H]
  _Float16* WHT  = (_Float16*)carve((size_t)H * N * 2);  // Wh^T f16  [H][N]
  float*    SV   = (float*)   carve((size_t)N * 4);
  float*    DV   = (float*)   carve((size_t)N * 4);
  _Float16* P    = (_Float16*)carve((size_t)N * N * 2);  // attention f16 [N][N]
  float*    HP   = (float*)   carve((size_t)N * H * 4);  // relu(P@Wh) f32
  float*    MU   = (float*)   carve((size_t)H * 4);
  float*    RSTD = (float*)   carve((size_t)H * 4);
  _Float16* HB16 = (_Float16*)carve((size_t)N * H * 2);  // BN output f16

  // 32x32 tile per wave: (4096/32)*(256/32) = 1024 tiles -> 1024 waves
  const int GEMM_BLOCKS = 128;  // 128 blocks * 8 waves = 1024 waves

  // ---- input conversions ----
  k_cvt_f16<<<(N * F) / 256, 256, 0, stream>>>(XH, x, N * F);
  k_transpose_f16<<<(F * H) / 256, 256, 0, stream>>>(W1T, W1, F, H);
  k_transpose_f16<<<(H * H) / 256, 256, 0, stream>>>(W2T, W2, H, H);

  // ================= Layer 1 =================
  k_gemm32<<<GEMM_BLOCKS, 256, 0, stream>>>(XH, W1T, WH, N, H, F, 0);
  k_rowdot2<<<N, 256, 0, stream>>>(WH, a1s, a1d, SV, DV, H);
  k_transpose_f16<<<(N * H) / 256, 256, 0, stream>>>(WHT, WH, N, H);
  k_attn_row<<<N, 256, 0, stream>>>(SV, DV, adj, P);
  k_gemm32<<<GEMM_BLOCKS, 256, 0, stream>>>(P, WHT, HP, N, H, N, 1);
  k_colstats<<<H, 256, 0, stream>>>(HP, MU, RSTD, N, H);
  k_bn_f16<<<(N * H) / 256, 256, 0, stream>>>(HP, MU, RSTD, gamma1, beta1, HB16, N * H, H);

  // ================= Layer 2 =================
  k_gemm32<<<GEMM_BLOCKS, 256, 0, stream>>>(HB16, W2T, WH, N, H, H, 0);
  k_rowdot2<<<N, 256, 0, stream>>>(WH, a2s, a2d, SV, DV, H);
  k_transpose_f16<<<(N * H) / 256, 256, 0, stream>>>(WHT, WH, N, H);
  k_attn_row<<<N, 256, 0, stream>>>(SV, DV, adj, P);
  k_gemm32<<<GEMM_BLOCKS, 256, 0, stream>>>(P, WHT, HP, N, H, N, 1);
  k_colstats<<<H, 256, 0, stream>>>(HP, MU, RSTD, N, H);
  k_bn_f32<<<(N * H) / 256, 256, 0, stream>>>(HP, MU, RSTD, gamma2, beta2, out, N * H, H);
}
